// T5CrossAttention_48653389529383
// MI455X (gfx1250) — matrix operand order repaired and checked by reading, not compiled, not run on hardware
//
#include <hip/hip_runtime.h>
#include <hip/hip_fp16.h>

typedef __attribute__((ext_vector_type(16))) _Float16 v16h;
typedef __attribute__((ext_vector_type(8)))  float    v8f;

union H16 { v16h h; uint4 u[2]; _Float16 e[16]; };
union H8  { uint4 u; _Float16 e[8]; };

__device__ __forceinline__ v8f zero8() {
  v8f z = {0.f,0.f,0.f,0.f,0.f,0.f,0.f,0.f};
  return z;
}

__device__ __forceinline__ v8f wmma_f16(v16h a, v16h b, v8f c) {
  // (neg_a, A, neg_b, B, c_mod, C, reuse_a, reuse_b)
  return __builtin_amdgcn_wmma_f32_16x16x32_f16(false, a, false, b, (short)0, c,
                                                false, false);
}

// CDNA5 async copy: 16B per lane, global -> LDS, tracked by ASYNCcnt.
// LDS operand is the wave-relative byte offset (low 32 bits of the flat
// address of a __shared__ object).
__device__ __forceinline__ void async_copy16(void* lds_dst, const void* gsrc) {
  unsigned loff = (unsigned)(unsigned long long)lds_dst;
  asm volatile("global_load_async_to_lds_b128 %0, %1, off"
               :: "v"(loff), "v"(gsrc) : "memory");
}
__device__ __forceinline__ void wait_async0() {
  asm volatile("s_wait_asynccnt 0x0" ::: "memory");
}

// ---------------------------------------------------------------------------
// K convert: f32 -> f16, flat copy. 8 elements / thread.
// ---------------------------------------------------------------------------
__global__ __launch_bounds__(256)
void convert_k_kernel(const float* __restrict__ Kg, _Float16* __restrict__ Kh) {
  size_t i = ((size_t)blockIdx.x * 256 + threadIdx.x) * 8;
  float4 a = *(const float4*)(Kg + i);
  float4 b = *(const float4*)(Kg + i + 4);
  H8 r;
  r.e[0] = (_Float16)a.x; r.e[1] = (_Float16)a.y;
  r.e[2] = (_Float16)a.z; r.e[3] = (_Float16)a.w;
  r.e[4] = (_Float16)b.x; r.e[5] = (_Float16)b.y;
  r.e[6] = (_Float16)b.z; r.e[7] = (_Float16)b.w;
  *(uint4*)(Kh + i) = r.u;
}

// ---------------------------------------------------------------------------
// V transpose-convert: [bh][kv][d] f32 -> [bh][d][kv] f16 via LDS 64x64 tile.
// grid: (BH=64, KVL/64=32), block 256.
// ---------------------------------------------------------------------------
__global__ __launch_bounds__(256)
void convert_v_kernel(const float* __restrict__ Vg, _Float16* __restrict__ Vt) {
  constexpr int HD = 64, KVL = 2048;
  __shared__ __align__(16) _Float16 T[64][64];   // [d][kv]
  const int bh  = blockIdx.x;
  const int kv0 = blockIdx.y * 64;
  const float* src = Vg + ((size_t)bh * KVL + kv0) * HD;
  for (int it = 0; it < 4; ++it) {
    int s   = threadIdx.x + 256 * it;   // 0..1023 segments of 4 floats
    int row = s >> 4;                   // kv row, 16 segs per row
    int cs  = (s & 15) * 4;             // d
    float4 v = *(const float4*)(src + (size_t)row * HD + cs);
    T[cs + 0][row] = (_Float16)v.x;
    T[cs + 1][row] = (_Float16)v.y;
    T[cs + 2][row] = (_Float16)v.z;
    T[cs + 3][row] = (_Float16)v.w;
  }
  __syncthreads();
  for (int it = 0; it < 2; ++it) {
    int s  = threadIdx.x + 256 * it;    // 0..511 segments of 8 halves
    int d  = s >> 3;                    // 8 segs per row
    int cs = (s & 7) * 8;
    *(uint4*)(Vt + ((size_t)bh * HD + d) * KVL + kv0 + cs) =
        *(const uint4*)&T[d][cs];
  }
}

// ---------------------------------------------------------------------------
// GEMM: C(4096x1024) = A(4096x1024) * B(1024x1024)
//   A_IS_F16:   A operand already stored as _Float16 (else f32, convert)
//   OUT_TO_QWS: scatter f16 result into [B,H,QL,HD] layout, else f32 row-major
// Block: 256 thr (8 waves), block tile 128x128, wave tile 64x32, K-step 32.
// ---------------------------------------------------------------------------
template<bool A_IS_F16, bool OUT_TO_QWS>
__global__ __launch_bounds__(256)
void gemm128_kernel(const void* __restrict__ Araw,
                    const float* __restrict__ Bmat,
                    void* __restrict__ Out) {
  constexpr int Nsz = 1024, Ksz = 1024;
  __shared__ __align__(16) _Float16 Al[128][32];   // [m][k]
  __shared__ __align__(16) _Float16 Bt[128][32];   // transposed: [n][k]

  const int tid  = threadIdx.x;
  const int lane = tid & 31;
  const int w    = tid >> 5;     // 0..7
  const int wm   = w >> 2;       // 0..1
  const int wn   = w & 3;        // 0..3
  const int lr   = lane & 15;
  const int hi   = lane >> 4;
  const int nBase = blockIdx.x * 128;
  const int mBase = blockIdx.y * 128;

  v8f acc[4][2];
  for (int s = 0; s < 4; ++s)
    for (int t = 0; t < 2; ++t) acc[s][t] = zero8();

  for (int k0 = 0; k0 < Ksz; k0 += 32) {
    // ---- stage A tile (128x32) into LDS as f16 ----
    if constexpr (A_IS_F16) {
      const _Float16* A = (const _Float16*)Araw;
      for (int it = 0; it < 4; ++it) {
        int s   = tid + 256 * it;          // 0..1023
        int row = s >> 3;
        int cs  = (s & 7) * 4;             // 4 halves per segment
        uint2 v = *(const uint2*)(A + (size_t)(mBase + row) * Ksz + k0 + cs);
        *(uint2*)&Al[row][cs] = v;
      }
    } else {
      const float* A = (const float*)Araw;
      for (int it = 0; it < 4; ++it) {
        int s   = tid + 256 * it;
        int row = s >> 3;
        int cs  = (s & 7) * 4;
        float4 v = *(const float4*)(A + (size_t)(mBase + row) * Ksz + k0 + cs);
        Al[row][cs + 0] = (_Float16)v.x;
        Al[row][cs + 1] = (_Float16)v.y;
        Al[row][cs + 2] = (_Float16)v.z;
        Al[row][cs + 3] = (_Float16)v.w;
      }
    }
    // ---- stage B tile (32x128) transposed into LDS as f16 ----
    for (int it = 0; it < 4; ++it) {
      int s   = tid + 256 * it;
      int row = s >> 5;                    // k 0..31
      int cs  = (s & 31) * 4;              // n 0..124
      float4 v = *(const float4*)(Bmat + (size_t)(k0 + row) * Nsz + nBase + cs);
      Bt[cs + 0][row] = (_Float16)v.x;
      Bt[cs + 1][row] = (_Float16)v.y;
      Bt[cs + 2][row] = (_Float16)v.z;
      Bt[cs + 3][row] = (_Float16)v.w;
    }
    __syncthreads();

    // ---- fragments (one ds clause) + WMMA chain ----
    H16 af[4], bf[2];
    for (int s = 0; s < 4; ++s) {
      int row = wm * 64 + s * 16 + lr;
      af[s].u[0] = *(const uint4*)&Al[row][hi * 8];
      af[s].u[1] = *(const uint4*)&Al[row][16 + hi * 8];
    }
    for (int t = 0; t < 2; ++t) {
      int col = wn * 32 + t * 16 + lr;
      bf[t].u[0] = *(const uint4*)&Bt[col][hi * 8];
      bf[t].u[1] = *(const uint4*)&Bt[col][16 + hi * 8];
    }
    for (int s = 0; s < 4; ++s)
      for (int t = 0; t < 2; ++t)
        acc[s][t] = wmma_f16(af[s].h, bf[t].h, acc[s][t]);
    __syncthreads();
  }

  // ---- epilogue ----
  for (int s = 0; s < 4; ++s)
    for (int t = 0; t < 2; ++t) {
      int c = nBase + wn * 32 + t * 16 + lr;
      for (int j = 0; j < 8; ++j) {
        int r = mBase + wm * 64 + s * 16 + j + 8 * hi;
        float val = acc[s][t][j];
        if constexpr (OUT_TO_QWS) {
          int b = r >> 10, q = r & 1023;   // QL = 1024
          int h = c >> 6,  d = c & 63;     // HD = 64
          ((_Float16*)Out)[(((size_t)b * 16 + h) * 1024 + q) * 64 + d] =
              (_Float16)val;
        } else {
          ((float*)Out)[(size_t)r * 1024 + c] = val;
        }
      }
    }
}

// ---------------------------------------------------------------------------
// Flash attention: one block per (b*H+h, 64 query rows); 4 waves x 16 rows.
// K/V pre-converted to f16 (V transposed) -> staging is pure async byte copy
// with double buffering. f16 WMMA, f32 accumulators, online softmax.
// NOTE: T5 attention has no 1/sqrt(d) scaling.
// ---------------------------------------------------------------------------
__global__ __launch_bounds__(128)
void flash_attn_kernel(const _Float16* __restrict__ Qws,
                       const _Float16* __restrict__ Kh,   // [bh][kv][d]
                       const _Float16* __restrict__ Vt,   // [bh][d][kv]
                       const float* __restrict__ maskp,
                       _Float16* __restrict__ AttnOut) {
  constexpr int HD = 64, KVL = 2048, QL = 1024;
  __shared__ __align__(16) _Float16 Kl[2][64][64];   // [buf][kv][d]
  __shared__ __align__(16) _Float16 Vl[2][64][64];   // [buf][d][kv]
  __shared__ __align__(16) _Float16 Pl[4][16][64];   // per-wave P staging

  const int tid  = threadIdx.x;
  const int lane = tid & 31;
  const int w    = tid >> 5;             // 0..3
  const int lr   = lane & 15;
  const int hi   = lane >> 4;
  const int bh   = blockIdx.x;           // b*16 + h
  const int b    = bh >> 4;
  const int h    = bh & 15;
  const int q0   = blockIdx.y * 64;

  const _Float16* Qbase = Qws + ((size_t)bh * QL + q0) * HD;
  const _Float16* Kbase = Kh + (size_t)bh * KVL * HD;   // tile rows contiguous
  const _Float16* Vbase = Vt + (size_t)bh * HD * KVL;   // row stride KVL
  const float*    mbase = maskp + (size_t)b * KVL;

  // Issue one 64x64 f16 K tile (8KB contiguous) + V tile (64 rows of 128B)
  auto issue_tile = [&](int buf, int kv0) {
    const char* ks = (const char*)(Kbase + (size_t)kv0 * HD);
    char* kd = (char*)&Kl[buf][0][0];
    for (int i = 0; i < 4; ++i) {
      int off = (i * 128 + tid) * 16;
      async_copy16(kd + off, ks + off);
    }
    for (int i = 0; i < 4; ++i) {
      int idx = i * 128 + tid;           // 512 segments of 16B
      int d   = idx >> 3;
      int seg = idx & 7;
      async_copy16((char*)&Vl[buf][d][seg * 8],
                   (const char*)(Vbase + (size_t)d * KVL + kv0 + seg * 8));
    }
  };

  // A-fragments of this wave's 16 query rows (K = HD split into 2x32)
  H16 qa[2];
  {
    const _Float16* qp = Qbase + (size_t)(w * 16 + lr) * HD;
    for (int f = 0; f < 2; ++f) {
      qa[f].u[0] = *(const uint4*)(qp + f * 32 + hi * 8);
      qa[f].u[1] = *(const uint4*)(qp + f * 32 + 16 + hi * 8);
    }
  }

  v8f o[4];
  for (int n = 0; n < 4; ++n) o[n] = zero8();
  float m_prev[8], lsum[8];
  for (int j = 0; j < 8; ++j) { m_prev[j] = -3.0e38f; lsum[j] = 0.f; }

  issue_tile(0, 0);
  wait_async0();
  __syncthreads();

  int cur = 0;
  for (int kv0 = 0; kv0 < KVL; kv0 += 64) {
    if (kv0 + 64 < KVL) issue_tile(cur ^ 1, kv0 + 64);   // prefetch next

    // ---- hoist all K B-fragments (one ds clause), then 8 WMMAs ----
    H16 kb[4][2];
    for (int n = 0; n < 4; ++n) {
      const _Float16* kp = &Kl[cur][n * 16 + lr][0];
      for (int f = 0; f < 2; ++f) {
        kb[n][f].u[0] = *(const uint4*)(kp + f * 32 + hi * 8);
        kb[n][f].u[1] = *(const uint4*)(kp + f * 32 + 16 + hi * 8);
      }
    }
    v8f sacc[4];
    for (int n = 0; n < 4; ++n) {
      v8f c = zero8();
      c = wmma_f16(qa[0].h, kb[n][0].h, c);
      c = wmma_f16(qa[1].h, kb[n][1].h, c);
      float mv = mbase[kv0 + n * 16 + lr];
      for (int j = 0; j < 8; ++j) c[j] += mv;
      sacc[n] = c;
    }

    // ---- online softmax (row stats; xor 1/2/4/8 stays in 16-lane half) ----
    float m_new[8], alpha[8], rs[8];
    for (int j = 0; j < 8; ++j) {
      float mx = sacc[0][j];
      for (int n = 1; n < 4; ++n) mx = fmaxf(mx, sacc[n][j]);
      for (int d = 1; d < 16; d <<= 1) mx = fmaxf(mx, __shfl_xor(mx, d, 32));
      m_new[j] = fmaxf(m_prev[j], mx);
      alpha[j] = __expf(m_prev[j] - m_new[j]);
      rs[j] = 0.f;
    }
    for (int n = 0; n < 4; ++n)
      for (int j = 0; j < 8; ++j) {
        float p = __expf(sacc[n][j] - m_new[j]);
        sacc[n][j] = p;
        rs[j] += p;
      }
    for (int j = 0; j < 8; ++j) {
      for (int d = 1; d < 16; d <<= 1) rs[j] += __shfl_xor(rs[j], d, 32);
      lsum[j] = lsum[j] * alpha[j] + rs[j];
      m_prev[j] = m_new[j];
    }
    for (int n = 0; n < 4; ++n)
      for (int j = 0; j < 8; ++j) o[n][j] *= alpha[j];

    // ---- P: C-layout -> LDS -> A-layout (wave-local round trip) ----
    for (int n = 0; n < 4; ++n)
      for (int j = 0; j < 8; ++j)
        Pl[w][j + 8 * hi][n * 16 + lr] = (_Float16)sacc[n][j];
    asm volatile("s_wait_dscnt 0" ::: "memory");   // wave-local LDS RAW
    H16 pa[2];
    {
      const _Float16* pp = &Pl[w][lr][0];
      for (int f = 0; f < 2; ++f) {
        pa[f].u[0] = *(const uint4*)(pp + f * 32 + hi * 8);
        pa[f].u[1] = *(const uint4*)(pp + f * 32 + 16 + hi * 8);
      }
    }

    // ---- hoist V B-fragments, then O += P V ----
    H16 vb[4][2];
    for (int n = 0; n < 4; ++n) {
      const _Float16* vp = &Vl[cur][n * 16 + lr][0];
      for (int f = 0; f < 2; ++f) {
        vb[n][f].u[0] = *(const uint4*)(vp + f * 32 + hi * 8);
        vb[n][f].u[1] = *(const uint4*)(vp + f * 32 + 16 + hi * 8);
      }
    }
    for (int n = 0; n < 4; ++n) {
      o[n] = wmma_f16(pa[0].h, vb[n][0].h, o[n]);
      o[n] = wmma_f16(pa[1].h, vb[n][1].h, o[n]);
    }

    wait_async0();       // my async loads for next tile have landed
    __syncthreads();     // everyone's have landed; cur buffer free
    cur ^= 1;
  }

  // ---- normalize + store f16 to attn workspace [b*QL+q][h*64+d] ----
  for (int n = 0; n < 4; ++n) {
    int dg = h * 64 + n * 16 + lr;
    for (int j = 0; j < 8; ++j) {
      int qg = q0 + w * 16 + j + 8 * hi;
      float val = o[n][j] / lsum[j];
      AttnOut[((size_t)b * QL + qg) * 1024 + dg] = (_Float16)val;
    }
  }
}

// ---------------------------------------------------------------------------
extern "C" void kernel_launch(void* const* d_in, const int* in_sizes, int n_in,
                              void* d_out, int out_size, void* d_ws,
                              size_t ws_size, hipStream_t stream) {
  const float* hidden = (const float*)d_in[0];  // [4,1024,1024]
  const float* key    = (const float*)d_in[1];  // [4,16,2048,64]
  const float* value  = (const float*)d_in[2];  // [4,16,2048,64]
  const float* mask   = (const float*)d_in[3];  // [4,1,1,2048]
  const float* Wq     = (const float*)d_in[4];  // [1024,1024]
  const float* Wo     = (const float*)d_in[5];  // [1024,1024]
  float* out = (float*)d_out;                   // [4,1024,1024]

  // workspace layout (f16): qws 8MB | aws 8MB | Kh 16MB | Vt 16MB
  _Float16* qws = (_Float16*)d_ws;
  _Float16* aws = qws + (size_t)4 * 16 * 1024 * 64;
  _Float16* Kh  = aws + (size_t)4 * 1024 * 1024;
  _Float16* Vt  = Kh + (size_t)4 * 16 * 2048 * 64;

  // 0) pre-convert K (straight) and V (transposed) to f16
  {
    size_t kelems = (size_t)4 * 16 * 2048 * 64;      // 8M
    convert_k_kernel<<<(unsigned)(kelems / (256 * 8)), 256, 0, stream>>>(key, Kh);
    convert_v_kernel<<<dim3(64, 2048 / 64), 256, 0, stream>>>(value, Vt);
  }

  dim3 ggrid(1024 / 128, 4096 / 128);  // (N tiles, M tiles)

  // 1) Q projection -> f16 workspace in [B,H,QL,HD]
  gemm128_kernel<false, true><<<ggrid, 256, 0, stream>>>(hidden, Wq, (void*)qws);

  // 2) flash attention -> f16 workspace in [B*QL, H*HD]
  flash_attn_kernel<<<dim3(4 * 16, 1024 / 64), 128, 0, stream>>>(
      qws, Kh, Vt, mask, aws);

  // 3) output projection -> f32 d_out
  gemm128_kernel<true, false><<<ggrid, 256, 0, stream>>>((const void*)aws, Wo,
                                                         (void*)out);
}
